// ScaleDotProductAttention_4767413699214
// MI455X (gfx1250) — matrix operand order repaired and checked by reading, not compiled, not run on hardware
//
#include <hip/hip_runtime.h>
#include <hip/hip_bf16.h>
#include <hip/hip_fp16.h>

typedef __attribute__((ext_vector_type(16))) _Float16 v16h;
typedef __attribute__((ext_vector_type(8)))  float    v8f;

#define B_   4
#define H_   16
#define S_   2048
#define DK_  64
#define NEGV (-1e9f)
#define WAVES 4          // waves per block
#define QT    16         // q rows per wave

union AF { v16h v; _Float16 h[16]; };
union CF { v8f  v; float    f[8];  };

struct Smem {
    _Float16 s[WAVES][QT][S_];   // 4 * 16 * 2048 * 2B = 256 KB (of 320 KB/WGP)
    float    rowmax[WAVES][QT];
    float    invl[WAVES][QT];
};

__global__ __launch_bounds__(WAVES * 32)
void attn_fwd_wmma(const float* __restrict__ Q, const float* __restrict__ K,
                   const float* __restrict__ V, const int* __restrict__ mask,
                   float* __restrict__ out, float* __restrict__ probs)
{
    extern __shared__ char smem_raw[];
    Smem* sm = reinterpret_cast<Smem*>(smem_raw);

    const int lane = threadIdx.x & 31;
    const int w    = threadIdx.x >> 5;
    const int nq   = lane & 15;   // N / M-within-group index
    const int g    = lane >> 4;   // half-wave group

    const int qblocks = S_ / (WAVES * QT);          // 32
    const int bh = blockIdx.x / qblocks;            // 0..63
    const int q0 = (blockIdx.x % qblocks) * (WAVES * QT) + w * QT;

    const float scale = 0.125f;                     // 1/sqrt(64)

    const float* Qb = Q + (size_t)bh * S_ * DK_;
    const float* Kb = K + (size_t)bh * S_ * DK_;
    const float* Vb = V + (size_t)bh * S_ * DK_;
    float* outb   = out   + (size_t)bh * S_ * DK_;
    float* probsb = probs + (size_t)bh * S_ * S_;

    // ---- Q A-fragments (two 16x32 f16 tiles covering DK=64), loaded once ----
    // A layout (ISA 16-bit A 16x32): M = lane%16, K(j) = (j/8)*16 + g*8 + (j%8)
    AF qa[2];
#pragma unroll
    for (int dc = 0; dc < 2; ++dc) {
#pragma unroll
        for (int j = 0; j < 16; ++j) {
            int d = dc * 32 + ((j >> 3) << 4) + g * 8 + (j & 7);
            qa[dc].h[j] = (_Float16)Qb[(size_t)(q0 + nq) * DK_ + d];
        }
    }

    // =========================== Pass A: scores ============================
    float rm[8];
#pragma unroll
    for (int r = 0; r < 8; ++r) rm[r] = -3.0e38f;

    for (int kt = 0; kt < S_ / 16; ++kt) {
        if (kt + 1 < S_ / 16)   // global_prefetch_b8 next K tile
            __builtin_prefetch(Kb + (size_t)((kt + 1) * 16 + nq) * DK_ + g * 32, 0, 1);

        // B layout (32x16): N = lane%16 (k-row), K(j) = (j/8)*16 + g*8 + (j%8) (d index)
        AF kb[2];
        const int krow = kt * 16 + nq;
#pragma unroll
        for (int dc = 0; dc < 2; ++dc)
#pragma unroll
            for (int j = 0; j < 16; ++j)
                kb[dc].h[j] = (_Float16)Kb[(size_t)krow * DK_ + dc * 32 + ((j >> 3) << 4) + g * 8 + (j & 7)];

        v8f z = {};
        CF c;
        c.v = __builtin_amdgcn_wmma_f32_16x16x32_f16(false, qa[0].v, false, kb[0].v,
                                                     (short)0, z,   false, false);
        c.v = __builtin_amdgcn_wmma_f32_16x16x32_f16(false, qa[1].v, false, kb[1].v,
                                                     (short)0, c.v, false, false);

        const int kcol = kt * 16 + nq;
#pragma unroll
        for (int r = 0; r < 8; ++r) {           // D layout: M = r + 8g, N = lane%16
            int M = r + 8 * g;
            float s = c.f[r] * scale;
            if (mask[(size_t)(q0 + M) * S_ + kcol] == 0) s = NEGV;
            rm[r] = fmaxf(rm[r], s);
            sm->s[w][M][kcol] = (_Float16)s;
        }
    }

    // row-max reduction across the 16 lanes sharing a half-group
#pragma unroll
    for (int r = 0; r < 8; ++r) {
#pragma unroll
        for (int m = 1; m < 16; m <<= 1)
            rm[r] = fmaxf(rm[r], __shfl_xor(rm[r], m, 32));
    }
    if (nq == 0) {
#pragma unroll
        for (int r = 0; r < 8; ++r) sm->rowmax[w][r + 8 * g] = rm[r];
    }
    __syncthreads();

    // ====================== Pass B: softmax + P@V ==========================
    const float mrow = sm->rowmax[w][nq];   // this lane's A-row max
    CF o[4];
#pragma unroll
    for (int c4 = 0; c4 < 4; ++c4) { v8f z = {}; o[c4].v = z; }
    float lsum = 0.f;

    for (int kt = 0; kt < S_ / 32; ++kt) {
        if (kt + 1 < S_ / 32)   // prefetch next V tile
            __builtin_prefetch(Vb + (size_t)((kt + 1) * 32 + g * 16) * DK_ + nq, 0, 1);

        // P as A-fragment, straight out of LDS (transpose for free)
        AF pa;
#pragma unroll
        for (int j = 0; j < 16; ++j) {
            int k = kt * 32 + ((j >> 3) << 4) + g * 8 + (j & 7);
            float s = (float)sm->s[w][nq][k];
            float p = __expf(s - mrow);
            lsum += p;
            pa.h[j] = (_Float16)p;
            sm->s[w][nq][k] = (_Float16)p;   // stash unnormalized prob
        }
#pragma unroll
        for (int c4 = 0; c4 < 4; ++c4) {
            AF vb;                            // V B-fragment: K = (j/8)*16+g*8+(j%8), N = d col
#pragma unroll
            for (int j = 0; j < 16; ++j) {
                int vr = kt * 32 + ((j >> 3) << 4) + g * 8 + (j & 7);
                vb.h[j] = (_Float16)Vb[(size_t)vr * DK_ + c4 * 16 + nq];
            }
            o[c4].v = __builtin_amdgcn_wmma_f32_16x16x32_f16(false, pa.v, false, vb.v,
                                                             (short)0, o[c4].v,
                                                             false, false);
        }
    }

    // combine the two half-group partial row sums; 1/l per row
    lsum += __shfl_xor(lsum, 16, 32);
    if (lane < 16) sm->invl[w][lane] = 1.0f / lsum;
    __syncthreads();

    // ---------------- out = (P@V) / l ----------------
#pragma unroll
    for (int c4 = 0; c4 < 4; ++c4)
#pragma unroll
        for (int r = 0; r < 8; ++r) {
            int M = r + 8 * g;
            outb[(size_t)(q0 + M) * DK_ + c4 * 16 + nq] = o[c4].f[r] * sm->invl[w][M];
        }

    // ---------------- probs = p_hat / l, written once, coalesced ----------
    for (int idx = lane; idx < QT * S_; idx += 32) {
        int row = idx >> 11;            // /2048
        int col = idx & (S_ - 1);
        probsb[(size_t)(q0 + row) * S_ + col] =
            (float)sm->s[w][row][col] * sm->invl[w][row];
    }
}

extern "C" void kernel_launch(void* const* d_in, const int* in_sizes, int n_in,
                              void* d_out, int out_size, void* d_ws, size_t ws_size,
                              hipStream_t stream)
{
    (void)in_sizes; (void)n_in; (void)out_size; (void)d_ws; (void)ws_size;

    const float* Q    = (const float*)d_in[0];
    const float* K    = (const float*)d_in[1];
    const float* V    = (const float*)d_in[2];
    const int*   mask = (const int*)  d_in[3];

    float* out   = (float*)d_out;                                  // [B,H,S,DK]
    float* probs = out + (size_t)B_ * H_ * S_ * DK_;               // [B,H,S,S]

    dim3 grid(B_ * H_ * (S_ / (WAVES * QT)));   // 64 bh * 32 q-blocks = 2048
    dim3 block(WAVES * 32);                     // 128 threads = 4 waves
    size_t shmem = sizeof(Smem);                // ~256.5 KB of the 320 KB WGP LDS

    attn_fwd_wmma<<<grid, block, shmem, stream>>>(Q, K, V, mask, out, probs);
}
